// StreamGQASelfAttention2DRelPos_1632087573173
// MI455X (gfx1250) — matrix operand order repaired and checked by reading, not compiled, not run on hardware
//
#include <hip/hip_runtime.h>

// ---------------------------------------------------------------------------
// MI455X (gfx1250) StreamGQA attention, round 2.
//  - All heavy streams pre-converted to bf16 (halves HBM bytes; caches dominate).
//  - Weights pre-transposed to [N][K] bf16 so WMMA B-fragments are contiguous.
//  - Fragments loaded with 16B vector ops (global_load_b128 / ds_load_b128)
//    instead of per-element b32 gathers.
//  - Attention: block-cooperative LDS staging of K/V tiles (8 waves share),
//    flash-style online softmax, bias via pt/pf lookup tables (pf hoisted).
// ---------------------------------------------------------------------------

typedef __attribute__((ext_vector_type(16))) __bf16 v16bf;
typedef __attribute__((ext_vector_type(8)))  __bf16 bf8;   // 16 B
typedef __attribute__((ext_vector_type(4)))  __bf16 bf4;   // 8 B
typedef __attribute__((ext_vector_type(8)))  float  v8f;
typedef __attribute__((ext_vector_type(4)))  float  f4;

#define DEV static __device__ __forceinline__
#define LDP 72   // padded LDS row stride in halves (multiple of 8 -> 16B align)

DEV v8f wmma_bf16(v16bf a, v16bf b, v8f c) {
  return __builtin_amdgcn_wmma_f32_16x16x32_bf16(false, a, false, b,
                                                 (short)0, c, false, false);
}

// A fragment (16x32 MxK) from bf16 row-major [m][k], ld in halves (mult of 8).
// lane L: m = L&15, k run = [16*(L>>4), 16*(L>>4)+16) contiguous.
DEV v16bf a_frag_bf(const __bf16* p, int ld) {
  const int L = threadIdx.x & 31;
  const __bf16* q = p + (L & 15) * ld + ((L >> 4) << 4);
  bf8 lo = *(const bf8*)q;
  bf8 hi = *(const bf8*)(q + 8);
  v16bf r;
#pragma unroll
  for (int h = 0; h < 8; ++h) { r[h] = lo[h]; r[h + 8] = hi[h]; }
  return r;
}

// B fragment (32x16 KxN) from bf16 stored [n][k] (transposed), ld in halves.
// lane L: n = L&15, k run = [16*(L>>4), 16*(L>>4)+16) contiguous.
DEV v16bf b_fragT_bf(const __bf16* p, int ld) {
  const int L = threadIdx.x & 31;
  const __bf16* q = p + (L & 15) * ld + ((L >> 4) << 4);
  bf8 lo = *(const bf8*)q;
  bf8 hi = *(const bf8*)(q + 8);
  v16bf r;
#pragma unroll
  for (int h = 0; h < 8; ++h) { r[h] = lo[h]; r[h + 8] = hi[h]; }
  return r;
}

// B fragment from fp32 stored [n][k] with row guard (n < nvalid), ld in floats.
DEV v16bf b_fragT_f32g(const float* p, int ld, int nvalid) {
  const int L = threadIdx.x & 31;
  const int n = L & 15;
  v16bf r;
  if (n < nvalid) {
    const f4* q = (const f4*)(p + n * ld + ((L >> 4) << 4));
    f4 x0 = q[0], x1 = q[1], x2 = q[2], x3 = q[3];
#pragma unroll
    for (int h = 0; h < 4; ++h) {
      r[h] = (__bf16)x0[h]; r[4 + h] = (__bf16)x1[h];
      r[8 + h] = (__bf16)x2[h]; r[12 + h] = (__bf16)x3[h];
    }
  } else {
#pragma unroll
    for (int h = 0; h < 16; ++h) r[h] = (__bf16)0.f;
  }
  return r;
}

// C/D fragment (16x16). lane L: n = L&15 ; m = 8*(L>>4) + r
template <typename F>
DEV void store_c_frag(const v8f& c, F f) {
  const int L = threadIdx.x & 31;
  const int n = L & 15;
  const int mb = (L >> 4) << 3;
#pragma unroll
  for (int r = 0; r < 8; ++r) f(mb + r, n, c[r]);
}

// ---------------------------------------------------------------------------
// fp32 -> bf16 convert (same layout), 4-wide
// ---------------------------------------------------------------------------
__global__ __launch_bounds__(256) void k_cvt(const float* __restrict__ s,
                                             __bf16* __restrict__ d, int n4) {
  int i = blockIdx.x * blockDim.x + threadIdx.x;
  if (i < n4) {
    f4 x = ((const f4*)s)[i];
    bf4 y;
#pragma unroll
    for (int u = 0; u < 4; ++u) y[u] = (__bf16)x[u];
    ((bf4*)d)[i] = y;
  }
}

// ---------------------------------------------------------------------------
// fp32 W[K][N] -> bf16 WT[N][K] (LDS tiled transpose), K,N multiples of 64
// ---------------------------------------------------------------------------
__global__ __launch_bounds__(256) void k_cvtT(const float* __restrict__ W,
                                              __bf16* __restrict__ WT,
                                              int K, int N) {
  __shared__ __bf16 t[64][65];
  const int n0 = blockIdx.x * 64, k0 = blockIdx.y * 64;
  const int tx = threadIdx.x & 63, ty = threadIdx.x >> 6;
#pragma unroll
  for (int i = 0; i < 16; ++i) {
    int k = ty + i * 4;
    t[k][tx] = (__bf16)W[(k0 + k) * N + n0 + tx];
  }
  __syncthreads();
#pragma unroll
  for (int i = 0; i < 16; ++i) {
    int n = ty + i * 4;
    WT[(n0 + n) * K + k0 + tx] = t[tx][n];
  }
}

// ---------------------------------------------------------------------------
// Kernel: qs[bh][n][d] = 0.125 * (q @ Wq)  (bf16). grid (16,16), block 128.
// ---------------------------------------------------------------------------
__global__ __launch_bounds__(128) void k_proj_q(const __bf16* __restrict__ qb,
                                                const __bf16* __restrict__ wqT,
                                                __bf16* __restrict__ qs) {
  const int wave = threadIdx.x >> 5;
  const int m0 = blockIdx.x * 64 + wave * 16;
  const int c0 = blockIdx.y * 64;
  v8f acc[4] = {};
  for (int kk = 0; kk < 1024; kk += 32) {
    v16bf a = a_frag_bf(qb + m0 * 1024 + kk, 1024);
#pragma unroll
    for (int t = 0; t < 4; ++t) {
      v16bf b = b_fragT_bf(wqT + (c0 + t * 16) * 1024 + kk, 1024);
      acc[t] = wmma_bf16(a, b, acc[t]);
    }
  }
#pragma unroll
  for (int t = 0; t < 4; ++t)
    store_c_frag(acc[t], [&](int m, int n, float v) {
      int row = m0 + m, b = row >> 9, ns = row & 511;
      int col = c0 + t * 16 + n, h = col >> 6, d = col & 63;
      qs[(((b << 4) + h) * 512 + ns) * 64 + d] = (__bf16)(v * 0.125f);
    });
}

// ---------------------------------------------------------------------------
// Kernel: K/V projections. z=0: kp_t[bg][d][n] ; z=1: vp[bg][n][d]
// grid (16,4,2), block 128
// ---------------------------------------------------------------------------
__global__ __launch_bounds__(128) void k_proj_kv(
    const __bf16* __restrict__ kb_, const __bf16* __restrict__ vb_,
    const __bf16* __restrict__ wkT, const __bf16* __restrict__ wvT,
    __bf16* __restrict__ kp_bf, float* __restrict__ kp_f,
    __bf16* __restrict__ vp_bf, float* __restrict__ vp_f) {
  const bool isV = blockIdx.z != 0;
  const __bf16* __restrict__ src = isV ? vb_ : kb_;
  const __bf16* __restrict__ WT = isV ? wvT : wkT;
  const int wave = threadIdx.x >> 5;
  const int m0 = blockIdx.x * 64 + wave * 16;
  const int c0 = blockIdx.y * 64;
  v8f acc[4] = {};
  for (int kk = 0; kk < 1024; kk += 32) {
    v16bf a = a_frag_bf(src + m0 * 1024 + kk, 1024);
#pragma unroll
    for (int t = 0; t < 4; ++t) {
      v16bf b = b_fragT_bf(WT + (c0 + t * 16) * 1024 + kk, 1024);
      acc[t] = wmma_bf16(a, b, acc[t]);
    }
  }
#pragma unroll
  for (int t = 0; t < 4; ++t)
    store_c_frag(acc[t], [&](int m, int n, float v) {
      int row = m0 + m, b = row >> 9, ns = row & 511;
      int col = c0 + t * 16 + n, g = col >> 6, d = col & 63;
      int bg = (b << 2) + g;
      if (!isV) {
        int idx = (bg * 64 + d) * 512 + ns;
        kp_bf[idx] = (__bf16)v; kp_f[idx] = v;
      } else {
        int idx = (bg * 512 + ns) * 64 + d;
        vp_bf[idx] = (__bf16)v; vp_f[idx] = v;
      }
    });
}

// ---------------------------------------------------------------------------
// Kernel: pt[16384][256] = qs @ emb_t^T (255 valid rows of emb_t)
// grid (256,4), block 128
// ---------------------------------------------------------------------------
__global__ __launch_bounds__(128) void k_bias_t(const __bf16* __restrict__ qs,
                                                const float* __restrict__ emb_t,
                                                float* __restrict__ pt) {
  const int wave = threadIdx.x >> 5;
  const int m0 = blockIdx.x * 64 + wave * 16;
  const int c0 = blockIdx.y * 64;
  v8f acc[4] = {};
#pragma unroll
  for (int kk = 0; kk < 64; kk += 32) {
    v16bf a = a_frag_bf(qs + m0 * 64 + kk, 64);
#pragma unroll
    for (int t = 0; t < 4; ++t) {
      int r0 = c0 + t * 16;
      v16bf b = b_fragT_f32g(emb_t + r0 * 64 + kk, 64, 255 - r0);
      acc[t] = wmma_bf16(a, b, acc[t]);
    }
  }
#pragma unroll
  for (int t = 0; t < 4; ++t)
    store_c_frag(acc[t], [&](int m, int n, float v) {
      pt[(m0 + m) * 256 + c0 + t * 16 + n] = v;
    });
}

// ---------------------------------------------------------------------------
// Kernel: pf[16384][16] = qs @ emb_f^T (15 valid rows). grid (256), block 128
// ---------------------------------------------------------------------------
__global__ __launch_bounds__(128) void k_bias_f(const __bf16* __restrict__ qs,
                                                const float* __restrict__ emb_f,
                                                float* __restrict__ pf) {
  const int wave = threadIdx.x >> 5;
  const int m0 = blockIdx.x * 64 + wave * 16;
  v8f acc = {};
#pragma unroll
  for (int kk = 0; kk < 64; kk += 32) {
    v16bf a = a_frag_bf(qs + m0 * 64 + kk, 64);
    v16bf b = b_fragT_f32g(emb_f + kk, 64, 15);
    acc = wmma_bf16(a, b, acc);
  }
  store_c_frag(acc, [&](int m, int n, float v) {
    pf[(m0 + m) * 16 + n] = v;
  });
}

// ---------------------------------------------------------------------------
// Flash attention per head. grid (4, 32), block 256 (8 waves).
// Block stages 64x64 bf16 K/V tiles into LDS (transposed, padded stride);
// each wave owns 16 query rows: 8 WMMA (S) + 8 WMMA (P@V) per 64-col tile.
// ---------------------------------------------------------------------------
__global__ __launch_bounds__(256) void k_attn(
    const __bf16* __restrict__ ckb, const __bf16* __restrict__ cvb,
    const __bf16* __restrict__ qs, const __bf16* __restrict__ kp_bf,
    const __bf16* __restrict__ vp_bf, const float* __restrict__ pt,
    const float* __restrict__ pf, __bf16* __restrict__ ao) {
  __shared__ __bf16 ktile[64 * LDP];        // [j][d]
  __shared__ __bf16 vtile[64 * LDP];        // [d][j]
  __shared__ __bf16 pbuf[8][16 * LDP];      // per-wave P, [m][j]
  const int tid = threadIdx.x;
  const int wave = tid >> 5, L = tid & 31, cn = L & 15, half = L >> 4;
  const int bh = blockIdx.y;
  const int bg = ((bh >> 4) << 2) + ((bh & 15) >> 2);
  const int row0 = blockIdx.x * 128 + wave * 16;

  // Q fragments (loaded once)
  const __bf16* qrow = qs + (bh * 512 + row0) * 64;
  v16bf aq[2];
#pragma unroll
  for (int p = 0; p < 2; ++p) aq[p] = a_frag_bf(qrow + p * 32, 64);

  // hoisted bias terms: i = row0 + half*8 + r ; i>>3 = row0>>3 + half ; i&7 = r
  float pfv[8];
#pragma unroll
  for (int r = 0; r < 8; ++r)
    pfv[r] = pf[(bh * 512 + row0 + half * 8 + r) * 16 + (r - (cn & 7) + 7)];
  const float* ptbase = pt + (bh * 512 + row0 + half * 8) * 256;
  const int Cdt = 192 + (row0 >> 3) + half;

  v8f o[4] = {};
  float rmax[8], rsum[8];
#pragma unroll
  for (int r = 0; r < 8; ++r) { rmax[r] = -1e30f; rsum[r] = 0.f; }

  for (int j0 = 0; j0 < 2048; j0 += 64) {
    const bool cached = j0 < 1536;
    const __bf16* ksrc; int kld;
    const __bf16* vsrc;
    if (cached) {
      ksrc = ckb + (size_t)bh * 64 * 1536 + j0; kld = 1536;
      vsrc = cvb + ((size_t)bh * 1536 + j0) * 64;
    } else {
      ksrc = kp_bf + (size_t)bg * 64 * 512 + (j0 - 1536); kld = 512;
      vsrc = vp_bf + ((size_t)bg * 512 + (j0 - 1536)) * 64;
    }
    // prefetch next K tile
    if (j0 + 64 < 2048)
      __builtin_prefetch((const char*)(ksrc + 64) + (size_t)tid * 64, 0, 3);

    // ---- stage K (transpose [d][j] -> [j][d]) and V ([j][d] -> [d][j]) ----
#pragma unroll
    for (int c = 0; c < 2; ++c) {
      int e = (tid + c * 256) * 8;          // half index in 64x64 tile
      int r_ = e >> 6, c_ = e & 63;
      bf8 x = *(const bf8*)(ksrc + r_ * kld + c_);   // K row d=r_, cols j=c_..c_+7
#pragma unroll
      for (int u = 0; u < 8; ++u) ktile[(c_ + u) * LDP + r_] = x[u];
      bf8 y = *(const bf8*)(vsrc + r_ * 64 + c_);    // V row j=r_, cols d=c_..c_+7
#pragma unroll
      for (int u = 0; u < 8; ++u) vtile[(c_ + u) * LDP + r_] = y[u];
    }
    __syncthreads();

    // ---- S = Q @ K ----
    v8f s[4];
#pragma unroll
    for (int jt = 0; jt < 4; ++jt) {
      v8f acc = {};
#pragma unroll
      for (int p = 0; p < 2; ++p)
        acc = wmma_bf16(aq[p], b_fragT_bf(ktile + jt * 16 * LDP + p * 32, LDP),
                        acc);
      s[jt] = acc;
    }

    // ---- bias ----
#pragma unroll
    for (int jt = 0; jt < 4; ++jt) {
      int dt = Cdt - ((j0 >> 3) + jt * 2 + (cn >> 3));
      int rt = (dt > 127 ? 127 : dt) + 127;
#pragma unroll
      for (int r = 0; r < 8; ++r)
        s[jt][r] += ptbase[r * 256 + rt] + pfv[r];
    }

    // ---- online softmax (rows live in 16-lane halves) ----
    float p_[4][8];
#pragma unroll
    for (int r = 0; r < 8; ++r) {
      float mx = s[0][r];
#pragma unroll
      for (int jt = 1; jt < 4; ++jt) mx = fmaxf(mx, s[jt][r]);
#pragma unroll
      for (int off = 1; off < 16; off <<= 1)
        mx = fmaxf(mx, __shfl_xor(mx, off, 32));
      float nmax = fmaxf(rmax[r], mx);
      float scale = __expf(rmax[r] - nmax);
      float lsum = 0.f;
#pragma unroll
      for (int jt = 0; jt < 4; ++jt) {
        float pv = __expf(s[jt][r] - nmax);
        p_[jt][r] = pv;
        lsum += pv;
      }
#pragma unroll
      for (int off = 1; off < 16; off <<= 1)
        lsum += __shfl_xor(lsum, off, 32);
      rsum[r] = rsum[r] * scale + lsum;
      rmax[r] = nmax;
#pragma unroll
      for (int t = 0; t < 4; ++t) o[t][r] *= scale;
    }

    // ---- P: C-layout -> A-layout via per-wave LDS ----
    __bf16* pb = pbuf[wave];
#pragma unroll
    for (int jt = 0; jt < 4; ++jt)
#pragma unroll
      for (int r = 0; r < 8; ++r)
        pb[(half * 8 + r) * LDP + jt * 16 + cn] = (__bf16)p_[jt][r];
    v16bf pa[2];
#pragma unroll
    for (int p = 0; p < 2; ++p) pa[p] = a_frag_bf(pb + p * 32, LDP);

    // ---- O += P @ V ----
#pragma unroll
    for (int t = 0; t < 4; ++t)
#pragma unroll
      for (int p = 0; p < 2; ++p)
        o[t] = wmma_bf16(pa[p], b_fragT_bf(vtile + t * 16 * LDP + p * 32, LDP),
                         o[t]);
    __syncthreads();
  }

  // ---- normalize, store bf16 ao[bh][n][d] ----
#pragma unroll
  for (int t = 0; t < 4; ++t)
    store_c_frag(o[t], [&](int m, int n, float v) {
      ao[(bh * 512 + row0 + m) * 64 + t * 16 + n] = (__bf16)(v / rsum[m & 7]);
    });
}

// ---------------------------------------------------------------------------
// Kernel: out = ao_perm @ Wo. grid (16,16), block 128
// ---------------------------------------------------------------------------
__global__ __launch_bounds__(128) void k_out(const __bf16* __restrict__ ao,
                                             const __bf16* __restrict__ woT,
                                             float* __restrict__ out) {
  const int wave = threadIdx.x >> 5;
  const int m0 = blockIdx.x * 64 + wave * 16;
  const int c0 = blockIdx.y * 64;
  const int b = m0 >> 9, ns0 = m0 & 511;
  v8f acc[4] = {};
  for (int kk = 0; kk < 1024; kk += 32) {
    int h = kk >> 6;
    v16bf a = a_frag_bf(ao + (((b << 4) + h) * 512 + ns0) * 64 + (kk & 63), 64);
#pragma unroll
    for (int t = 0; t < 4; ++t) {
      v16bf bfr = b_fragT_bf(woT + (c0 + t * 16) * 1024 + kk, 1024);
      acc[t] = wmma_bf16(a, bfr, acc[t]);
    }
  }
#pragma unroll
  for (int t = 0; t < 4; ++t)
    store_c_frag(acc[t], [&](int m, int n, float v) {
      out[(m0 + m) * 1024 + c0 + t * 16 + n] = v;
    });
}

// ---------------------------------------------------------------------------
// Kernel: fp32 cache shift+append (vectorized), GQA-replicated
// ---------------------------------------------------------------------------
__global__ __launch_bounds__(256) void k_cache(
    const float* __restrict__ ck, const float* __restrict__ cv,
    const float* __restrict__ kpf, const float* __restrict__ vpf,
    float* __restrict__ outk, float* __restrict__ outv) {
  const int total4 = 32 * 64 * 1536 / 4;
  for (int i4 = blockIdx.x * blockDim.x + threadIdx.x; i4 < total4;
       i4 += gridDim.x * blockDim.x) {
    int e = i4 * 4;
    int bh = e / 98304, rem = e - bh * 98304;
    int bg = ((bh >> 4) << 2) + ((bh & 15) >> 2);
    int d = rem / 1536, c = rem % 1536;
    f4 vk = (c < 1024) ? *(const f4*)&ck[(bh * 64 + d) * 1536 + 512 + c]
                       : *(const f4*)&kpf[(bg * 64 + d) * 512 + (c - 1024)];
    ((f4*)outk)[i4] = vk;
    int j = rem / 64, dd = rem % 64;
    f4 vv = (j < 1024) ? *(const f4*)&cv[(bh * 1536 + 512 + j) * 64 + dd]
                       : *(const f4*)&vpf[(bg * 512 + (j - 1024)) * 64 + dd];
    ((f4*)outv)[i4] = vv;
  }
}

// ---------------------------------------------------------------------------
extern "C" void kernel_launch(void* const* d_in, const int* in_sizes, int n_in,
                              void* d_out, int out_size, void* d_ws,
                              size_t ws_size, hipStream_t stream) {
  const float* q  = (const float*)d_in[0];
  const float* k  = (const float*)d_in[1];
  const float* v  = (const float*)d_in[2];
  const float* ck = (const float*)d_in[3];
  const float* cv = (const float*)d_in[4];
  const float* Wq = (const float*)d_in[5];
  const float* Wk = (const float*)d_in[6];
  const float* Wv = (const float*)d_in[7];
  const float* Wo = (const float*)d_in[8];
  const float* emb_t = (const float*)d_in[9];
  const float* emb_f = (const float*)d_in[10];

  char* ws = (char*)d_ws;
  const size_t MB = 1u << 20;
  __bf16* qb    = (__bf16*)(ws);                  // 2 MB
  __bf16* kb_   = (__bf16*)(ws + 2 * MB);         // 2 MB
  __bf16* vb_   = (__bf16*)(ws + 4 * MB);         // 2 MB
  __bf16* wqT   = (__bf16*)(ws + 6 * MB);         // 2 MB
  __bf16* wkT   = (__bf16*)(ws + 8 * MB);         // 0.5 MB
  __bf16* wvT   = (__bf16*)(ws + 8 * MB + 512 * 1024);
  __bf16* woT   = (__bf16*)(ws + 9 * MB);         // 2 MB
  __bf16* qs_bf = (__bf16*)(ws + 11 * MB);        // 2 MB
  __bf16* kp_bf = (__bf16*)(ws + 13 * MB);        // 0.5 MB
  __bf16* vp_bf = (__bf16*)(ws + 13 * MB + 512 * 1024);
  float*  kp_f  = (float*)(ws + 14 * MB);         // 1 MB
  float*  vp_f  = (float*)(ws + 15 * MB);         // 1 MB
  __bf16* ckb   = (__bf16*)(ws + 16 * MB);        // 6 MB
  __bf16* cvb   = (__bf16*)(ws + 22 * MB);        // 6 MB
  float*  pt    = (float*)(ws + 28 * MB);         // 16 MB
  float*  pf    = (float*)(ws + 44 * MB);         // 1 MB
  __bf16* ao_bf = (__bf16*)(ws + 45 * MB);        // 2 MB

  float* outp = (float*)d_out;
  float* outk = outp + 2 * 512 * 1024;
  float* outv = outk + 32 * 64 * 1536;

  // precision conversions / weight transposes
  k_cvt<<<1024, 256, 0, stream>>>(q, qb, 262144);
  k_cvt<<<1024, 256, 0, stream>>>(k, kb_, 262144);
  k_cvt<<<1024, 256, 0, stream>>>(v, vb_, 262144);
  k_cvt<<<3072, 256, 0, stream>>>(ck, ckb, 786432);
  k_cvt<<<3072, 256, 0, stream>>>(cv, cvb, 786432);
  k_cvtT<<<dim3(16, 16), 256, 0, stream>>>(Wq, wqT, 1024, 1024);
  k_cvtT<<<dim3(4, 16), 256, 0, stream>>>(Wk, wkT, 1024, 256);
  k_cvtT<<<dim3(4, 16), 256, 0, stream>>>(Wv, wvT, 1024, 256);
  k_cvtT<<<dim3(16, 16), 256, 0, stream>>>(Wo, woT, 1024, 1024);

  // projections + bias tables
  k_proj_q<<<dim3(16, 16), 128, 0, stream>>>(qb, wqT, qs_bf);
  k_proj_kv<<<dim3(16, 4, 2), 128, 0, stream>>>(kb_, vb_, wkT, wvT, kp_bf,
                                                kp_f, vp_bf, vp_f);
  k_bias_t<<<dim3(256, 4), 128, 0, stream>>>(qs_bf, emb_t, pt);
  k_bias_f<<<dim3(256), 128, 0, stream>>>(qs_bf, emb_f, pf);

  // attention + output projection + cache update
  k_attn<<<dim3(4, 32), 256, 0, stream>>>(ckb, cvb, qs_bf, kp_bf, vp_bf, pt,
                                          pf, ao_bf);
  k_out<<<dim3(16, 16), 128, 0, stream>>>(ao_bf, woT, outp);
  k_cache<<<1024, 256, 0, stream>>>(ck, cv, kp_f, vp_f, outk, outv);
}
